// ComboActionScorer_79182017069087
// MI455X (gfx1250) — matrix operand order repaired and checked by reading, not compiled, not run on hardware
//
#include <hip/hip_runtime.h>

#define EMBED   256
#define STATE   128
#define HID     256
#define N_ACT   256
#define N_NODES 2048
#define TEAMS   8
#define PAIR_F  14
#define BT      512

#define ROWS    64          // rows (actions) per block in main kernel
#define MT      4           // 16-row M tiles per block

typedef __attribute__((ext_vector_type(16))) __bf16        v16bf;
typedef __attribute__((ext_vector_type(8)))  float         v8f;
typedef __attribute__((ext_vector_type(8)))  unsigned int  v8u;

__device__ inline unsigned short f2bf(float f) {
    // round-to-nearest-even float -> bf16 (bits)
    unsigned int u = __builtin_bit_cast(unsigned int, f);
    u += 0x7FFFu + ((u >> 16) & 1u);
    return (unsigned short)(u >> 16);
}

// ---------------- K0: global_emb = mean(node_embeddings, axis=0) ----------------
__global__ void k_global_mean(const float* __restrict__ ne, float* __restrict__ g) {
    int e = threadIdx.x;
    float s = 0.f;
    for (int n = 0; n < N_NODES; ++n) s += ne[n * EMBED + e];
    g[e] = s * (1.0f / (float)N_NODES);
}

// ---------------- K1: actA[a,h] = pe@W1p + te@W1t + g@W1g + b1 ----------------
__global__ void k_actA(const float* __restrict__ ne, const int* __restrict__ idx,
                       const float* __restrict__ team, const float* __restrict__ g,
                       const float* __restrict__ W1, const float* __restrict__ b1,
                       float* __restrict__ actA) {
    __shared__ float pe[EMBED], te[EMBED], gl[EMBED];
    int a = blockIdx.x, h = threadIdx.x;
    int node = idx[a];
    pe[h] = ne[node * EMBED + h];
    te[h] = team[(a % TEAMS) * EMBED + h];
    gl[h] = g[h];
    __syncthreads();
    float acc = b1[h];
    for (int e = 0; e < EMBED; ++e) {
        acc += pe[e] * W1[e * HID + h];
        acc += te[e] * W1[(EMBED + e) * HID + h];
        acc += gl[e] * W1[(2 * EMBED + e) * HID + h];
    }
    actA[a * HID + h] = acc;
}

// ---------------- K2: stateC[bt,h] = relu(states@Ws+bs) @ W1_state ----------------
__global__ void k_stateC(const float* __restrict__ states, const float* __restrict__ Ws,
                         const float* __restrict__ bs, const float* __restrict__ W1,
                         float* __restrict__ stateC) {
    __shared__ float st[STATE];
    __shared__ float se[EMBED];
    int bt = blockIdx.x, h = threadIdx.x;
    if (h < STATE) st[h] = states[bt * STATE + h];
    __syncthreads();
    float tmp = bs[h];
    for (int k = 0; k < STATE; ++k) tmp += st[k] * Ws[k * EMBED + h];
    se[h] = fmaxf(tmp, 0.f);
    __syncthreads();
    float acc = 0.f;
    for (int e = 0; e < EMBED; ++e) acc += se[e] * W1[(3 * EMBED + e) * HID + h];
    stateC[bt * HID + h] = acc;
}

// ---------------- K3: repack W2 into bf16 WMMA B-fragment order ----------------
// dword index d = (((ct*8 + ks)*32) + lane)*8 + j
// lane<16 : N=ct*16+lane,    K = ks*32 + 16*(j>>2) + 2*(j&3),(+1)
// lane>=16: N=ct*16+lane-16, K = ks*32 + 8 + 16*(j>>2) + 2*(j&3),(+1)
__global__ void k_w2frag(const float* __restrict__ W2, unsigned int* __restrict__ W2frag) {
    int d = blockIdx.x * blockDim.x + threadIdx.x;   // 0..32767
    int j    = d & 7;
    int lane = (d >> 3) & 31;
    int ks   = (d >> 8) & 7;
    int ct   = d >> 11;
    int N  = ct * 16 + (lane & 15);
    int K0 = ks * 32 + (lane >> 4) * 8 + (j >> 2) * 16 + 2 * (j & 3);
    unsigned int lo = f2bf(W2[K0 * HID + N]);
    unsigned int hi = f2bf(W2[(K0 + 1) * HID + N]);
    W2frag[d] = lo | (hi << 16);
}

// ---------------- K3b: repack W1_pair rows (K=14, padded to 32) into B-fragment order ----------------
// dword index d = (ct*32 + lane)*8 + j ; single K-step of 32
__global__ void k_w1pfrag(const float* __restrict__ W1, unsigned int* __restrict__ W1pfrag) {
    int d = blockIdx.x * blockDim.x + threadIdx.x;   // 0..4095
    int j    = d & 7;
    int lane = (d >> 3) & 31;
    int ct   = d >> 8;
    int N  = ct * 16 + (lane & 15);
    int K0 = (lane >> 4) * 8 + (j >> 2) * 16 + 2 * (j & 3);
    unsigned int lo = (K0     < PAIR_F) ? (unsigned int)f2bf(W1[(4 * EMBED + K0    ) * HID + N]) : 0u;
    unsigned int hi = (K0 + 1 < PAIR_F) ? (unsigned int)f2bf(W1[(4 * EMBED + K0 + 1) * HID + N]) : 0u;
    W1pfrag[d] = lo | (hi << 16);
}

__device__ inline v16bf load_bfrag(const unsigned int* __restrict__ p) {
    uint4 x = *(const uint4*)p;
    uint4 y = *(const uint4*)(p + 4);
    v8u r;
    r[0] = x.x; r[1] = x.y; r[2] = x.z; r[3] = x.w;
    r[4] = y.x; r[5] = y.y; r[6] = y.z; r[7] = y.w;
    return __builtin_bit_cast(v16bf, r);
}

// ---------------- K4: main fused kernel (64 rows per block) ----------------
__global__ void __launch_bounds__(256) k_main(
    const float* __restrict__ pair,
    const float* __restrict__ actA, const float* __restrict__ stateC,
    const unsigned int* __restrict__ W2frag, const unsigned int* __restrict__ W1pfrag,
    const float* __restrict__ b2, const float* __restrict__ W3,
    const float* __restrict__ b3, float* __restrict__ out)
{
    __shared__ unsigned short h1lds[ROWS][264];   // bf16, padded rows (132 dwords)
    __shared__ float pairlds[ROWS][PAIR_F];
    __shared__ float scorebuf[8][ROWS];

    int blk = blockIdx.x;            // 2048 blocks
    int bt  = blk >> 2;              // 0..511
    int a0  = (blk & 3) * ROWS;      // action tile base
    int t   = threadIdx.x;

    // cooperative load of the 64x14 pair-feature tile
    for (int i = t; i < ROWS * PAIR_F; i += 256) {
        int r = i / PAIR_F, f = i % PAIR_F;
        pairlds[r][f] = pair[(bt * N_ACT + a0 + r) * PAIR_F + f];
    }
    __syncthreads();

    int l = t & 31, w = t >> 5;
    int half = l >> 4, lm = l & 15;
    int ct0 = 2 * w, ct1 = 2 * w + 1;
    int n0 = ct0 * 16 + lm, n1 = ct1 * 16 + lm;

    // ---- layer 1: h1 = relu(actA + stateC + pair @ W1_pair), pair part on WMMA (K=14 padded to 32)
    {
        float sc0 = stateC[bt * HID + n0];
        float sc1 = stateC[bt * HID + n1];
        v16bf Bp0 = load_bfrag(W1pfrag + (ct0 * 32 + l) * 8);
        v16bf Bp1 = load_bfrag(W1pfrag + (ct1 * 32 + l) * 8);
        const v8f zero = {};
#pragma unroll
        for (int m = 0; m < MT; ++m) {
            // build pair A-fragment (ISA 16-bit 16x32 layout); K >= 14 zero-padded
            int row = 16 * m + lm;
            v8u ar;
#pragma unroll
            for (int j = 0; j < 4; ++j) {
                int K0 = half * 8 + 2 * j;
                unsigned int lo = (K0     < PAIR_F) ? (unsigned int)f2bf(pairlds[row][K0])     : 0u;
                unsigned int hi = (K0 + 1 < PAIR_F) ? (unsigned int)f2bf(pairlds[row][K0 + 1]) : 0u;
                ar[j] = lo | (hi << 16);
            }
            ar[4] = 0u; ar[5] = 0u; ar[6] = 0u; ar[7] = 0u;   // K in [16,32) all >= 14
            v16bf Ap = __builtin_bit_cast(v16bf, ar);

            v8f p0 = __builtin_amdgcn_wmma_f32_16x16x32_bf16(false, Ap, false, Bp0, (short)0, zero, false, false);
            v8f p1 = __builtin_amdgcn_wmma_f32_16x16x32_bf16(false, Ap, false, Bp1, (short)0, zero, false, false);

#pragma unroll
            for (int v = 0; v < 8; ++v) {
                int rv = 16 * m + half * 8 + v;
                float h0 = p0[v] + actA[(a0 + rv) * HID + n0] + sc0;
                float h1 = p1[v] + actA[(a0 + rv) * HID + n1] + sc1;
                h1lds[rv][n0] = f2bf(fmaxf(h0, 0.f));
                h1lds[rv][n1] = f2bf(fmaxf(h1, 0.f));
            }
        }
    }
    __syncthreads();

    // ---- layer 2: h2 = relu(h1 @ W2 + b2); B kept in registers across 4 M-tiles
    v8f acc[2 * MT] = {};
#pragma unroll
    for (int ks = 0; ks < 8; ++ks) {
        v16bf B0 = load_bfrag(W2frag + ((ct0 * 8 + ks) * 32 + l) * 8);
        v16bf B1 = load_bfrag(W2frag + ((ct1 * 8 + ks) * 32 + l) * 8);
        int ad = ks * 16 + half * 4;
#pragma unroll
        for (int m = 0; m < MT; ++m) {
            const unsigned int* rowp = (const unsigned int*)&h1lds[16 * m + lm][0];
            uint4 alo = *(const uint4*)(rowp + ad);
            uint4 ahi = *(const uint4*)(rowp + ad + 8);
            v8u arw;
            arw[0] = alo.x; arw[1] = alo.y; arw[2] = alo.z; arw[3] = alo.w;
            arw[4] = ahi.x; arw[5] = ahi.y; arw[6] = ahi.z; arw[7] = ahi.w;
            v16bf A = __builtin_bit_cast(v16bf, arw);
            acc[2 * m]     = __builtin_amdgcn_wmma_f32_16x16x32_bf16(false, A, false, B0, (short)0, acc[2 * m],     false, false);
            acc[2 * m + 1] = __builtin_amdgcn_wmma_f32_16x16x32_bf16(false, A, false, B1, (short)0, acc[2 * m + 1], false, false);
        }
    }

    // ---- layer 3: score = relu(h2) . W3 + b3, reduced in-register
    {
        float b20 = b2[n0], b21 = b2[n1];
        float w30 = W3[n0], w31 = W3[n1];
#pragma unroll
        for (int m = 0; m < MT; ++m) {
#pragma unroll
            for (int v = 0; v < 8; ++v) {
                float p = fmaxf(acc[2 * m][v] + b20, 0.f) * w30 +
                          fmaxf(acc[2 * m + 1][v] + b21, 0.f) * w31;
#pragma unroll
                for (int msk = 1; msk < 16; msk <<= 1)
                    p += __shfl_xor(p, msk, 32);
                if (lm == 0) scorebuf[w][m * 16 + half * 8 + v] = p;
            }
        }
    }
    __syncthreads();
    if (t < ROWS) {
        float s = b3[0];
#pragma unroll
        for (int ww = 0; ww < 8; ++ww) s += scorebuf[ww][t];
        out[bt * N_ACT + a0 + t] = s;   // [B,T,1,A] flat
    }
}

extern "C" void kernel_launch(void* const* d_in, const int* in_sizes, int n_in,
                              void* d_out, int out_size, void* d_ws, size_t ws_size,
                              hipStream_t stream) {
    const float* ne     = (const float*)d_in[0];
    const float* states = (const float*)d_in[1];
    const float* pair   = (const float*)d_in[2];
    const int*   idx    = (const int*)  d_in[3];
    const float* team   = (const float*)d_in[4];
    const float* Ws     = (const float*)d_in[5];
    const float* bs     = (const float*)d_in[6];
    const float* W1     = (const float*)d_in[7];
    const float* b1     = (const float*)d_in[8];
    const float* W2     = (const float*)d_in[9];
    const float* b2     = (const float*)d_in[10];
    const float* W3     = (const float*)d_in[11];
    const float* b3     = (const float*)d_in[12];
    float* out = (float*)d_out;

    float* ws      = (float*)d_ws;
    float* g       = ws;                        // 256 f32
    float* actA    = ws + 256;                  // 256*256 f32
    float* stateC  = actA + N_ACT * HID;        // 512*256 f32
    unsigned int* W2frag  = (unsigned int*)(stateC + BT * HID);  // 32768 dwords
    unsigned int* W1pfrag = W2frag + 32768;                      // 4096 dwords

    k_global_mean<<<1, 256, 0, stream>>>(ne, g);
    k_actA<<<N_ACT, 256, 0, stream>>>(ne, idx, team, g, W1, b1, actA);
    k_stateC<<<BT, 256, 0, stream>>>(states, Ws, bs, W1, stateC);
    k_w2frag<<<128, 256, 0, stream>>>(W2, W2frag);
    k_w1pfrag<<<16, 256, 0, stream>>>(W1, W1pfrag);
    k_main<<<BT * (N_ACT / ROWS), 256, 0, stream>>>(pair, actA, stateC, W2frag, W1pfrag,
                                                    b2, W3, b3, out);
}
